// AttentionModule_867583394307
// MI455X (gfx1250) — matrix operand order repaired and checked by reading, not compiled, not run on hardware
//
#include <hip/hip_runtime.h>

// ---------- types ----------
typedef __bf16         v16bf __attribute__((ext_vector_type(16)));
typedef __bf16         v2bf  __attribute__((ext_vector_type(2)));
typedef float          v8f   __attribute__((ext_vector_type(8)));
typedef float          v4f   __attribute__((ext_vector_type(4)));
typedef unsigned short v8u   __attribute__((ext_vector_type(8)));
typedef unsigned int   v4u32 __attribute__((ext_vector_type(4)));
typedef int            v8i32 __attribute__((ext_vector_type(8)));
typedef int            v4i32 __attribute__((ext_vector_type(4)));

union FragU  { v16bf bf; v8u h[2]; };
union Pack8  { v8u v; unsigned int u[4]; };

#define NPOS 4096
#define CDIM 512
#define CHD  256
#define NB   4

// padded LDS row strides (elements) -> conflict-free b128 fragment loads
#define QK_STR 264     // 256 + 8  (lane stride 132 dw == 4 mod 64)
#define VT_STR 72      // 64 + 8   (lane stride 36 dw  -> disjoint banks)
#define PB_STR 72
#define SB_STR 66      // f32 rows (stride 66 dw == 2 mod 64)
#define OB_STR 68      // f32 rows for epilogue staging (16B-aligned rows)

__device__ __forceinline__ unsigned short f32_bf16(float f) {
    return __builtin_bit_cast(unsigned short, (__bf16)f);   // fptrunc, RNE
}

__device__ __forceinline__ unsigned int pack_bf16x2(float lo, float hi) {
#if __has_builtin(__builtin_amdgcn_cvt_pk_bf16_f32)
    return __builtin_bit_cast(unsigned int, __builtin_amdgcn_cvt_pk_bf16_f32(lo, hi));
#else
    v2bf t;
    t[0] = (__bf16)lo;
    t[1] = (__bf16)hi;
    return __builtin_bit_cast(unsigned int, t);
#endif
}

__device__ __forceinline__ v16bf load_frag(const unsigned short* lo,
                                           const unsigned short* hi) {
    FragU f;
    f.h[0] = *(const v8u*)lo;   // ds_load_b128
    f.h[1] = *(const v8u*)hi;   // ds_load_b128
    return f.bf;
}

__device__ __forceinline__ v8f wmma_bf16(v16bf a, v16bf b, v8f c) {
    return __builtin_amdgcn_wmma_f32_16x16x32_bf16(false, a, false, b,
                                                   (short)0, c, false, false);
}

// ---------- TDM: 2D bf16 tile load with LDS row padding ----------
#define PAD_QK ((1u << 20) | (6u << 22) | (3u << 25))  // every 128 dw, +4 dw
#define PAD_VT ((1u << 20) | (4u << 22) | (3u << 25))  // every  32 dw, +4 dw

__device__ __forceinline__ void tdm_load_2d(unsigned int lds_off,
    const void* gaddr, unsigned int tensor_d0, unsigned int tensor_d1,
    unsigned int tile_d0, unsigned int tile_d1, unsigned long long stride0,
    unsigned int pad_bits)
{
    unsigned long long ga = (unsigned long long)(size_t)gaddr;
    v4u32 g0;
    g0[0] = 1u;                                              // count=1, user mode
    g0[1] = lds_off;                                         // LDS byte address
    g0[2] = (unsigned int)(ga & 0xFFFFFFFFull);              // global_addr lo
    g0[3] = (unsigned int)((ga >> 32) & 0x1FFFFFFull) | 0x80000000u; // hi | type=2
    v8i32 g1;
    g1[0] = (int)(0x10000u | pad_bits);                      // data_size=1 (2B) | pad
    g1[1] = (int)((tensor_d0 & 0xFFFFu) << 16);
    g1[2] = (int)((tensor_d0 >> 16) | ((tensor_d1 & 0xFFFFu) << 16));
    g1[3] = (int)((tensor_d1 >> 16) | (tile_d0 << 16));
    g1[4] = (int)tile_d1;                                    // tile_dim2 = 0
    g1[5] = (int)(stride0 & 0xFFFFFFFFull);
    g1[6] = (int)((stride0 >> 32) & 0xFFFFull);              // stride1 = 0
    g1[7] = 0;
    v4i32 z4 = {0, 0, 0, 0};
#if defined(__clang_major__) && (__clang_major__ >= 23)
    v8i32 z8 = {0, 0, 0, 0, 0, 0, 0, 0};
    __builtin_amdgcn_tensor_load_to_lds(g0, g1, z4, z4, z8, 0);
#else
    __builtin_amdgcn_tensor_load_to_lds(g0, g1, z4, z4, 0);
#endif
}

// ============================================================================
// Projection: y[o][n] = sum_c W[o][c] * X[b][c][n] + bias[o]   (bf16 out)
// store_t==0 : Y[b][o][n] (natural, for v)   store_t==1 : Y[b][n][o] (q, k)
// ============================================================================
__global__ __launch_bounds__(256) void proj_kernel(
    const float* __restrict__ X, const float* __restrict__ W,
    const float* __restrict__ bias, unsigned short* __restrict__ Y,
    int O, int store_t)
{
    __shared__ unsigned short Wt[64 * 40];   // [o][c] tile, padded row (32+8)
    __shared__ unsigned short Xt[64 * 40];   // [n][c] tile, padded row

    const int tid  = threadIdx.x;
    const int wave = tid >> 5, lane = tid & 31;
    const int l    = lane & 15, half = lane >> 4;
    const int wm   = wave >> 1, wn = wave & 1;
    const int n_blk = blockIdx.x * 64;
    const int o_blk = blockIdx.y * 64;
    const int b     = blockIdx.z;
    const float* Xb = X + (size_t)b * CDIM * NPOS;

    v8f acc[2];
    const v8f vzero = {0.f,0.f,0.f,0.f,0.f,0.f,0.f,0.f};
    acc[0] = vzero; acc[1] = vzero;

    for (int kk = 0; kk < 16; ++kk) {
        const int c0 = kk * 32;
        __syncthreads();
        {   // W tile: f32 -> bf16, [o][c]
            int o = tid >> 2, coff = (tid & 3) * 8;
            const float* wp = W + (size_t)(o_blk + o) * CDIM + c0 + coff;
            v4f w0 = *(const v4f*)wp;
            v4f w1 = *(const v4f*)(wp + 4);
            Pack8 pk;
            pk.u[0] = pack_bf16x2(w0[0], w0[1]);
            pk.u[1] = pack_bf16x2(w0[2], w0[3]);
            pk.u[2] = pack_bf16x2(w1[0], w1[1]);
            pk.u[3] = pack_bf16x2(w1[2], w1[3]);
            *(v8u*)(Wt + o * 40 + coff) = pk.v;
        }
        {   // X tile transposed: Xt[n][c] <- X[c0+c][n_blk+n]
            int n = tid & 63, cb = tid >> 6;
            #pragma unroll
            for (int i = 0; i < 8; ++i) {
                int c = cb + 4 * i;
                Xt[n * 40 + c] = f32_bf16(Xb[(size_t)(c0 + c) * NPOS + n_blk + n]);
            }
        }
        __syncthreads();

        const int ar = 16 * wm + l;
        v16bf a = load_frag(Wt + ar * 40 + 8 * half,
                            Wt + ar * 40 + 16 + 8 * half);
        #pragma unroll
        for (int t = 0; t < 2; ++t) {
            int br = 32 * wn + 16 * t + l;
            v16bf bb = load_frag(Xt + br * 40 + 8 * half,
                                 Xt + br * 40 + 16 + 8 * half);
            acc[t] = wmma_bf16(a, bb, acc[t]);
        }
    }

    float bv[8];
    #pragma unroll
    for (int r = 0; r < 8; ++r) bv[r] = bias[o_blk + 16 * wm + r + 8 * half];

    if (store_t) {          // Y[b][n][O]: one packed 16B store per tile per lane
        const size_t ybase = (size_t)b * NPOS * O;
        #pragma unroll
        for (int t = 0; t < 2; ++t) {
            int n = n_blk + 32 * wn + 16 * t + l;
            Pack8 pk;
            #pragma unroll
            for (int i = 0; i < 4; ++i)
                pk.u[i] = pack_bf16x2(acc[t][2*i] + bv[2*i], acc[t][2*i+1] + bv[2*i+1]);
            *(v8u*)(Y + ybase + (size_t)n * O + o_blk + 16 * wm + 8 * half) = pk.v;
        }
    } else {                // Y[b][O][n]
        const size_t ybase = (size_t)b * O * NPOS;
        #pragma unroll
        for (int t = 0; t < 2; ++t) {
            int n = n_blk + 32 * wn + 16 * t + l;
            #pragma unroll
            for (int r = 0; r < 8; ++r) {
                int o = o_blk + 16 * wm + r + 8 * half;
                Y[ybase + (size_t)o * NPOS + n] = f32_bf16(acc[t][r] + bv[r]);
            }
        }
    }
}

// ============================================================================
// Fused flash attention, TDM-staged (padded rows) + double buffered.
// ============================================================================
#define QT_OFF   0u                              // 64 x QK_STR bf16 = 33792 B
#define KT_OFF0  33792u                          // 2 buffers
#define KT_BUFSZ 33792u
#define VT_OFF0  101376u                         // 512 x VT_STR bf16 = 73728 B
#define VT_BUFSZ 73728u
#define SB_OFF   248832u                         // 64 x SB_STR f32 = 16896 B
#define PB_OFF   265728u                         // 64 x PB_STR bf16 = 9216 B
#define ROW_OFF  274944u                         // m/l/alpha rows (768 B)
#define SMEM_BYTES 275712u

__global__ __launch_bounds__(256) void attn_kernel(
    const unsigned short* __restrict__ qT,   // [B][N][256] bf16 (transposed)
    const unsigned short* __restrict__ kT,   // [B][N][256] bf16 (transposed)
    const unsigned short* __restrict__ vb,   // [B][512][N] bf16 (natural)
    const float* __restrict__ Vin,           // [B][512][N] f32 (residual)
    const float* __restrict__ gamma,
    float* __restrict__ out)                 // [B][512][N] f32
{
    extern __shared__ char smem[];
    unsigned short* Qt = (unsigned short*)(smem + QT_OFF);
    float*          Sb = (float*)(smem + SB_OFF);
    unsigned short* Pb = (unsigned short*)(smem + PB_OFF);
    float* mrow = (float*)(smem + ROW_OFF);
    float* lrow = mrow + 64;
    float* arow = lrow + 64;

    const int tid  = threadIdx.x;
    const int wave = tid >> 5, lane = tid & 31;
    const int l    = lane & 15, half = lane >> 4;
    const int wrow = wave & 3, wgrp = wave >> 2;
    const int b    = blockIdx.y;
    const int n0   = blockIdx.x * 64;

    const unsigned short* qTb = qT + (size_t)b * NPOS * CHD;
    const unsigned short* kTb = kT + (size_t)b * NPOS * CHD;
    const unsigned short* vbb = vb + (size_t)b * CDIM * NPOS;

    if (tid < 64) { mrow[tid] = -3.0e38f; lrow[tid] = 0.f; }

    // ---- TDM prefetch: Qt + (Kt,Vt) for block 0 ----
    if (wave == 0) {
        tdm_load_2d(QT_OFF,  qTb + (size_t)n0 * CHD, CHD, NPOS, CHD, 64, CHD, PAD_QK);
        tdm_load_2d(KT_OFF0, kTb,                    CHD, NPOS, CHD, 64, CHD, PAD_QK);
        tdm_load_2d(VT_OFF0, vbb,                    NPOS, CDIM, 64, CDIM, NPOS, PAD_VT);
    }

    v8f oacc[16];
    const v8f vzero = {0.f,0.f,0.f,0.f,0.f,0.f,0.f,0.f};
    #pragma unroll
    for (int t = 0; t < 16; ++t) oacc[t] = vzero;

    for (int j = 0; j < 64; ++j) {
        __syncthreads();   // readers of the buffer TDM will overwrite are done
        if (wave == 0) {
            if (j + 1 < 64) {
                const int m1 = (j + 1) * 64;
                const unsigned int bsel = (unsigned int)((j + 1) & 1);
                tdm_load_2d(KT_OFF0 + bsel * KT_BUFSZ, kTb + (size_t)m1 * CHD,
                            CHD, NPOS, CHD, 64, CHD, PAD_QK);
                tdm_load_2d(VT_OFF0 + bsel * VT_BUFSZ, vbb + m1,
                            NPOS, CDIM, 64, CDIM, NPOS, PAD_VT);
                __builtin_amdgcn_s_wait_tensorcnt(2);  // current buffers ready
            } else {
                __builtin_amdgcn_s_wait_tensorcnt(0);
            }
        }
        __syncthreads();   // staged tiles visible to all waves

        const unsigned short* Kt = (const unsigned short*)(smem + KT_OFF0 + (unsigned)(j & 1) * KT_BUFSZ);
        const unsigned short* Vt = (const unsigned short*)(smem + VT_OFF0 + (unsigned)(j & 1) * VT_BUFSZ);

        // ---- S = Q Kt^T : 64x64, K-reduce over 256 channels ----
        v8f sacc[2];
        sacc[0] = vzero; sacc[1] = vzero;
        #pragma unroll
        for (int ks = 0; ks < 8; ++ks) {
            const int c0 = ks * 32;
            const int ar = 16 * wrow + l;
            v16bf a = load_frag(Qt + ar * QK_STR + c0 + 8 * half,
                                Qt + ar * QK_STR + c0 + 16 + 8 * half);
            #pragma unroll
            for (int t = 0; t < 2; ++t) {
                int br = 32 * wgrp + 16 * t + l;
                v16bf bb = load_frag(Kt + br * QK_STR + c0 + 8 * half,
                                     Kt + br * QK_STR + c0 + 16 + 8 * half);
                sacc[t] = wmma_bf16(a, bb, sacc[t]);
            }
        }
        #pragma unroll
        for (int t = 0; t < 2; ++t) {
            int m = 32 * wgrp + 16 * t + l;
            #pragma unroll
            for (int r = 0; r < 8; ++r)
                Sb[(16 * wrow + r + 8 * half) * SB_STR + m] = sacc[t][r];
        }
        __syncthreads();

        // ---- online softmax: 4 threads per row, wave32 shfl reduction ----
        {
            int row = tid >> 2, q4 = tid & 3;
            const float* srow = Sb + row * SB_STR + q4 * 16;
            float mx = -3.0e38f;
            #pragma unroll
            for (int m = 0; m < 16; ++m) mx = fmaxf(mx, srow[m]);
            mx = fmaxf(mx, __shfl_xor(mx, 1));
            mx = fmaxf(mx, __shfl_xor(mx, 2));
            float mo_ = mrow[row];
            float mn = fmaxf(mo_, mx);
            float s = 0.f;
            unsigned int* prow = (unsigned int*)(Pb + row * PB_STR + q4 * 16);
            #pragma unroll
            for (int m = 0; m < 8; ++m) {
                float p0 = __expf(srow[2*m]   - mn);
                float p1 = __expf(srow[2*m+1] - mn);
                s += p0 + p1;
                prow[m] = pack_bf16x2(p0, p1);
            }
            s += __shfl_xor(s, 1);
            s += __shfl_xor(s, 2);
            if (q4 == 0) {
                float al = __expf(mo_ - mn);
                arow[row] = al;
                lrow[row] = lrow[row] * al + s;
                mrow[row] = mn;
            }
        }
        __syncthreads();

        // ---- rescale accumulators, then O += P * V^T ----
        float av[8];
        #pragma unroll
        for (int r = 0; r < 8; ++r) av[r] = arow[16 * wrow + r + 8 * half];
        #pragma unroll
        for (int t = 0; t < 16; ++t)
            #pragma unroll
            for (int r = 0; r < 8; ++r) oacc[t][r] *= av[r];

        #pragma unroll
        for (int ks = 0; ks < 2; ++ks) {
            const int mk = ks * 32;
            const int ar = 16 * wrow + l;
            v16bf a = load_frag(Pb + ar * PB_STR + mk + 8 * half,
                                Pb + ar * PB_STR + mk + 16 + 8 * half);
            #pragma unroll
            for (int t = 0; t < 16; ++t) {
                int c = 256 * wgrp + 16 * t + l;
                v16bf bb = load_frag(Vt + c * VT_STR + mk + 8 * half,
                                     Vt + c * VT_STR + mk + 16 + 8 * half);
                oacc[t] = wmma_bf16(a, bb, oacc[t]);
            }
        }
    }

    // ---- epilogue: stage gamma*O/l in LDS [c][n], then coalesced fused add ----
    float li[8];
    #pragma unroll
    for (int r = 0; r < 8; ++r) li[r] = 1.0f / lrow[16 * wrow + r + 8 * half];
    const float g = gamma[0];
    __syncthreads();                       // all staged tiles dead; reuse LDS
    float* Ob = (float*)smem;              // [512][OB_STR] = 139264 B
    #pragma unroll
    for (int t = 0; t < 16; ++t) {
        int c = 256 * wgrp + 16 * t + l;
        #pragma unroll
        for (int r = 0; r < 8; ++r)
            Ob[c * OB_STR + 16 * wrow + r + 8 * half] = g * oacc[t][r] * li[r];
    }
    __syncthreads();
    {
        const int cr = tid >> 2, nc = (tid & 3) * 16;
        #pragma unroll 2
        for (int i = 0; i < 8; ++i) {
            int c = cr + 64 * i;
            const float* obp = Ob + c * OB_STR + nc;
            size_t gidx = (size_t)(b * CDIM + c) * NPOS + n0 + nc;
            #pragma unroll
            for (int k = 0; k < 4; ++k) {
                v4f o = *(const v4f*)(obp + 4 * k);
                v4f v = *(const v4f*)(Vin + gidx + 4 * k);
                o = o + v;
                *(v4f*)(out + gidx + 4 * k) = o;
            }
        }
    }
}

// ============================================================================
extern "C" void kernel_launch(void* const* d_in, const int* in_sizes, int n_in,
                              void* d_out, int out_size, void* d_ws, size_t ws_size,
                              hipStream_t stream) {
    const float* Q    = (const float*)d_in[0];
    const float* K    = (const float*)d_in[1];
    const float* V    = (const float*)d_in[2];
    const float* Wq   = (const float*)d_in[3];
    const float* bq   = (const float*)d_in[4];
    const float* Wk   = (const float*)d_in[5];
    const float* bk   = (const float*)d_in[6];
    const float* Wv   = (const float*)d_in[7];
    const float* bv   = (const float*)d_in[8];
    const float* gmm  = (const float*)d_in[9];

    unsigned short* qws = (unsigned short*)d_ws;                     // [B][N][256]
    unsigned short* kws = qws + (size_t)NB * CHD * NPOS;             // [B][N][256]
    unsigned short* vws = kws + (size_t)NB * CHD * NPOS;             // [B][512][N]

    dim3 blk(256);
    proj_kernel<<<dim3(NPOS / 64, CHD / 64, NB), blk, 0, stream>>>(Q, Wq, bq, qws, CHD, 1);
    proj_kernel<<<dim3(NPOS / 64, CHD / 64, NB), blk, 0, stream>>>(K, Wk, bk, kws, CHD, 1);
    proj_kernel<<<dim3(NPOS / 64, CDIM / 64, NB), blk, 0, stream>>>(V, Wv, bv, vws, CDIM, 0);

    attn_kernel<<<dim3(NPOS / 64, NB), blk, SMEM_BYTES, stream>>>(
        qws, kws, vws, V, gmm, (float*)d_out);
}
